// SelfAttention_69647189672169
// MI455X (gfx1250) — matrix operand order repaired and checked by reading, not compiled, not run on hardware
//
#include <hip/hip_runtime.h>
#include <hip/hip_bf16.h>
#include <stdint.h>

// ---------------- problem constants ----------------
enum : int {
  HIDDEN = 2048,
  TSEQ   = 2048,
  NH     = 16,
  NKV    = 4,
  HD     = 128,
  KVW    = NKV * HD,   // 512
  QW     = NH * HD     // 2048
};

typedef __bf16 bf16_t;
typedef __bf16 v16bf __attribute__((ext_vector_type(16)));
typedef __bf16 bf16x8 __attribute__((ext_vector_type(8)));
typedef float  v8f   __attribute__((ext_vector_type(8)));
typedef int    v4i_vs __attribute__((vector_size(16)));   // matches builtin pointee

// ---- CDNA5 async global->LDS path (guarded; sync LDS fallback keeps semantics) ----
#if __has_builtin(__builtin_amdgcn_global_load_async_to_lds_b128)
#define HAVE_ASYNC_LDS 1
#else
#define HAVE_ASYNC_LDS 0
#endif

#if HAVE_ASYNC_LDS
#if __has_builtin(__builtin_amdgcn_s_wait_asynccnt)
#define WAIT_ASYNC(n) __builtin_amdgcn_s_wait_asynccnt(n)
#else
#define WAIT_ASYNC(n) asm volatile("s_wait_asynccnt %0" ::"i"(n) : "memory")
#endif
#else
#define WAIT_ASYNC(n)
#endif

__device__ __forceinline__ void stage_b128(const bf16_t* g, bf16_t* l) {
#if HAVE_ASYNC_LDS
  // signature (probe-confirmed): (AS1 v4i* src, AS3 v4i* dst, imm offset, imm cpol)
  __builtin_amdgcn_global_load_async_to_lds_b128(
      (__attribute__((address_space(1))) v4i_vs*)g,
      (__attribute__((address_space(3))) v4i_vs*)l, 0, 0);
#else
  *(bf16x8*)l = *(const bf16x8*)g;   // synchronous fallback, same data movement
#endif
}

__device__ __forceinline__ bf16_t f2bf(float f) {
  uint32_t u = __builtin_bit_cast(uint32_t, f);
  u += 0x7FFFu + ((u >> 16) & 1u);               // round-to-nearest-even
  uint16_t h = (uint16_t)(u >> 16);
  return __builtin_bit_cast(bf16_t, h);
}

// ---------------- elementwise kernels ----------------
__global__ void cast_f32_bf16(const float* __restrict__ src,
                              bf16_t* __restrict__ dst, int n) {
  int i = blockIdx.x * blockDim.x + threadIdx.x;
  if (i < n) dst[i] = f2bf(src[i]);
}

// src [K][N] f32  ->  dst [N][K] bf16
__global__ void tcast_f32_bf16(const float* __restrict__ src,
                               bf16_t* __restrict__ dst, int K, int N) {
  int i = blockIdx.x * blockDim.x + threadIdx.x;
  if (i >= K * N) return;
  int k = i / N, n = i - k * N;
  dst[(size_t)n * K + k] = f2bf(src[i]);
}

// reference rotate_half: x1 = even idx, x2 = odd idx, concat([-x2, x1])
__global__ void rope_cast(const float* __restrict__ src, bf16_t* __restrict__ dst,
                          int nheads, const float* __restrict__ rope_scale,
                          const int* __restrict__ pos_offset) {
  int i = blockIdx.x * blockDim.x + threadIdx.x;
  int total = TSEQ * nheads * HD;
  if (i >= total) return;
  int j  = i & (HD - 1);
  int th = i >> 7;                 // t*nheads + head
  int t  = th / nheads;
  int base = th * HD;
  float v = src[base + j];
  int jj = j & 63;
  float inv = expf(-0.14391156831212787f * (float)jj);   // 10000^(-jj/64)
  float ang = (float)(t + pos_offset[0]) * inv * rope_scale[0];
  float c = cosf(ang), s = sinf(ang);
  float partner = (j < 64) ? -src[base + 2 * j + 1] : src[base + 2 * (j - 64)];
  dst[base + j] = f2bf(v * c + partner * s);
}

// V f32 [T][NKV*HD] -> Vt bf16 [NKV*HD][T]
__global__ void vtrans(const float* __restrict__ vf, bf16_t* __restrict__ vt) {
  int i = blockIdx.x * blockDim.x + threadIdx.x;
  if (i >= TSEQ * KVW) return;
  int c = i & (KVW - 1);
  int t = i >> 9;
  vt[(size_t)c * TSEQ + t] = f2bf(vf[i]);
}

// ---------------- WMMA GEMM:  C[M][N] f32 = A[M][K]bf16 @ BT[N][K]bf16^T ----------------
// One block = 8 waves = 128(M) x 64(N) macro-tile. The 64x32 B panel per k-step is
// shared by all waves: staged once into LDS via async global->LDS DMA, double-buffered.
// A-frag and B-frag both use k = h*16 + i (contiguous order, same for A and B).
__global__ void __launch_bounds__(256) gemm_bf16_tn(
    const bf16_t* __restrict__ A, const bf16_t* __restrict__ BT,
    float* __restrict__ C, int M, int N, int K) {
  __shared__ bf16_t bstage[2][64 * 32];          // 2 x 4KB B panels
  int tid  = threadIdx.x;
  int wv   = tid >> 5;
  int lane = tid & 31;
  int l16  = lane & 15;
  int h    = lane >> 4;
  int ntiles = N >> 6;
  int mtb = blockIdx.x / ntiles;                 // 128-row macro block
  int nt  = blockIdx.x - mtb * ntiles;           // 64-col tile
  int mt  = (mtb << 3) + wv;                     // this wave's 16-row tile

  // staging: thread copies one 16B chunk: panel row = tid/4 (0..63), chunk = tid%4
  int srow = tid >> 2, schunk = tid & 3;
  const bf16_t* bsrc = BT + (size_t)((nt << 6) + srow) * K + schunk * 8;
  bf16_t* sdst0 = &bstage[0][srow * 32 + schunk * 8];
  bf16_t* sdst1 = &bstage[1][srow * 32 + schunk * 8];

  const bf16_t* arow = A + (size_t)((mt << 4) + l16) * K;
  v8f acc[4] = {};

  stage_b128(bsrc, sdst0);                       // prime buffer 0 (ASYNCcnt=1)
  int cur = 0;
  for (int kk = 0; kk < K; kk += 32) {
    bool more = (kk + 32) < K;
    if (more) stage_b128(bsrc + kk + 32, cur ? sdst0 : sdst1);  // prefetch next panel
    if (more) { WAIT_ASYNC(1); } else { WAIT_ASYNC(0); }        // current panel landed
    __syncthreads();                                            // visible to all waves

    __builtin_prefetch(arow + kk + 256, 0, 0);   // global_prefetch_b8 (A stream)
    v16bf af;
    {
      bf16x8 a0 = *(const bf16x8*)(arow + kk + h * 16);
      bf16x8 a1 = *(const bf16x8*)(arow + kk + h * 16 + 8);
#pragma unroll
      for (int i = 0; i < 8; ++i) { af[i] = a0[i]; af[8 + i] = a1[i]; }
    }
#pragma unroll
    for (int j = 0; j < 4; ++j) {
      const bf16_t* bp = &bstage[cur][(j * 16 + l16) * 32 + h * 16];
      v16bf bfr = *(const v16bf*)bp;             // 2x ds_load_b128
      acc[j] = __builtin_amdgcn_wmma_f32_16x16x32_bf16(
          false, af, false, bfr, (short)0, acc[j], false, false);
    }
    __syncthreads();                             // all reads done before re-staging
    cur ^= 1;
  }
#pragma unroll
  for (int j = 0; j < 4; ++j) {
    int n = (nt << 6) + (j << 4) + l16;
#pragma unroll
    for (int r = 0; r < 8; ++r) {
      int row = (mt << 4) + r + 8 * h;           // C layout: M = r + 8*(lane/16)
      C[(size_t)row * N + n] = acc[j][r];
    }
  }
}

// ---------------- flash attention (causal, GQA), one wave per (head, 16-query tile) ----
__global__ void __launch_bounds__(256) flash_attn(
    const bf16_t* __restrict__ Q, const bf16_t* __restrict__ Kc,
    const bf16_t* __restrict__ Vt, float* __restrict__ O) {
  __shared__ bf16_t pbuf[8][16 * 32];            // per-wave P re-layout staging
  int wv   = threadIdx.x >> 5;
  int lane = threadIdx.x & 31;
  int l16  = lane & 15;
  int h    = lane >> 4;
  int wave = blockIdx.x * 8 + wv;
  int head = wave >> 7;                          // 128 q-tiles per head
  int qt   = wave & 127;
  int kvh  = head >> 2;                          // GQA: 4 query heads share a KV head
  int qbase = qt << 4;

  const bf16_t* qrowp = Q + (size_t)(qbase + l16) * QW + head * HD;
  v16bf qf[4];
#pragma unroll
  for (int c = 0; c < 4; ++c) {
    bf16x8 a0 = *(const bf16x8*)(qrowp + c * 32 + h * 16);
    bf16x8 a1 = *(const bf16x8*)(qrowp + c * 32 + h * 16 + 8);
#pragma unroll
    for (int i = 0; i < 8; ++i) { qf[c][i] = a0[i]; qf[c][8 + i] = a1[i]; }
  }

  v8f o[8] = {};
  float m[8], l[8];
#pragma unroll
  for (int r = 0; r < 8; ++r) { m[r] = -__builtin_inff(); l[r] = 0.0f; }
  const float scale = 0.08838834764831845f;      // 1/sqrt(128)

  int nb = (qbase + 47) >> 5;                    // causal key blocks covering 0..qbase+15
  for (int b = 0; b < nb; ++b) {
    int kb = b << 5;
    int key0 = kb + l16, key1 = key0 + 16;
    const bf16_t* k0p = Kc + (size_t)key0 * KVW + kvh * HD;
    const bf16_t* k1p = Kc + (size_t)key1 * KVW + kvh * HD;
    v8f s0 = {}, s1 = {};
#pragma unroll
    for (int c = 0; c < 4; ++c) {                // S = Q @ K^T over d=128
      v16bf b0 = *(const v16bf*)(k0p + c * 32 + h * 16);
      s0 = __builtin_amdgcn_wmma_f32_16x16x32_bf16(false, qf[c], false, b0,
                                                   (short)0, s0, false, false);
      v16bf b1 = *(const v16bf*)(k1p + c * 32 + h * 16);
      s1 = __builtin_amdgcn_wmma_f32_16x16x32_bf16(false, qf[c], false, b1,
                                                   (short)0, s1, false, false);
    }
#pragma unroll
    for (int r = 0; r < 8; ++r) {                // scale + causal mask
      int rq = qbase + r + 8 * h;
      s0[r] = (key0 <= rq) ? s0[r] * scale : -3.0e38f;
      s1[r] = (key1 <= rq) ? s1[r] * scale : -3.0e38f;
    }
#pragma unroll
    for (int r = 0; r < 8; ++r) {                // online softmax per row
      float bm = fmaxf(s0[r], s1[r]);
#pragma unroll
      for (int msk = 1; msk < 16; msk <<= 1) bm = fmaxf(bm, __shfl_xor(bm, msk, 16));
      float mn = fmaxf(m[r], bm);
      float corr = expf(m[r] - mn);
      float p0 = expf(s0[r] - mn), p1 = expf(s1[r] - mn);
      float ps = p0 + p1;
#pragma unroll
      for (int msk = 1; msk < 16; msk <<= 1) ps += __shfl_xor(ps, msk, 16);
      l[r] = l[r] * corr + ps;
      m[r] = mn;
#pragma unroll
      for (int c = 0; c < 8; ++c) o[c][r] *= corr;
      int prow = r + 8 * h;                      // C-layout -> (row,col) in LDS
      pbuf[wv][prow * 32 + l16]      = f2bf(p0);
      pbuf[wv][prow * 32 + 16 + l16] = f2bf(p1);
    }
    asm volatile("s_wait_dscnt 0" ::: "memory"); // wave-local LDS store->load ordering
    v16bf pf;                                    // P as A-fragment (16x32)
    {
      const bf16_t* pr = &pbuf[wv][l16 * 32];
      bf16x8 a0 = *(const bf16x8*)(pr + h * 16);
      bf16x8 a1 = *(const bf16x8*)(pr + h * 16 + 8);
#pragma unroll
      for (int i = 0; i < 8; ++i) { pf[i] = a0[i]; pf[8 + i] = a1[i]; }
    }
#pragma unroll
    for (int c = 0; c < 8; ++c) {                // O += P @ V  (V^T rows contiguous)
      int dcol = c * 16 + l16;
      const bf16_t* vtp = Vt + (size_t)(kvh * HD + dcol) * TSEQ + kb + h * 16;
      v16bf bv = *(const v16bf*)vtp;
      o[c] = __builtin_amdgcn_wmma_f32_16x16x32_bf16(false, pf, false, bv,
                                                     (short)0, o[c], false, false);
    }
  }
#pragma unroll
  for (int c = 0; c < 8; ++c) {
    int dcol = c * 16 + l16;
#pragma unroll
    for (int r = 0; r < 8; ++r) {
      int rq = qbase + r + 8 * h;
      O[(size_t)rq * QW + head * HD + dcol] = o[c][r] / l[r];
    }
  }
}

// ---------------- host launch ----------------
extern "C" void kernel_launch(void* const* d_in, const int* in_sizes, int n_in,
                              void* d_out, int out_size, void* d_ws, size_t ws_size,
                              hipStream_t stream) {
  const float* x  = (const float*)d_in[0];
  // d_in[1] = mask (causal triu) — computed analytically in-kernel
  const float* Wq = (const float*)d_in[2];
  const float* Wk = (const float*)d_in[3];
  const float* Wv = (const float*)d_in[4];
  const float* Wo = (const float*)d_in[5];
  const float* rope_scale = (const float*)d_in[6];
  const int*   pos_offset = (const int*)d_in[7];

  char* w = (char*)d_ws;
  bf16_t* xb    = (bf16_t*)(w + 0);          //  8 MB  x bf16
  bf16_t* wqt   = (bf16_t*)(w + 8388608);    //  8 MB  Wq^T
  bf16_t* wkt   = (bf16_t*)(w + 16777216);   //  2 MB  Wk^T
  bf16_t* wvt   = (bf16_t*)(w + 18874368);   //  2 MB  Wv^T
  bf16_t* wot   = (bf16_t*)(w + 20971520);   //  8 MB  Wo^T
  float*  qf    = (float*) (w + 29360128);   // 16 MB  Q f32 (pre-rope)
  float*  kf    = (float*) (w + 46137344);   //  4 MB  K f32
  float*  vf    = (float*) (w + 50331648);   //  4 MB  V f32
  bf16_t* qr    = (bf16_t*)(w + 54525952);   //  8 MB  Q bf16 (post-rope)
  bf16_t* kr    = (bf16_t*)(w + 62914560);   //  2 MB  K bf16 (post-rope)
  bf16_t* vt    = (bf16_t*)(w + 65011712);   //  2 MB  V^T bf16
  float*  attnf = (float*) (w + 67108864);   // 16 MB  attention out f32
  bf16_t* attnb = (bf16_t*)(w + 83886080);   //  8 MB  attention out bf16
  if (ws_size < 92274688) return;            // need 88 MB scratch

  const int blk = 256;
  // 1) casts / transposes
  cast_f32_bf16<<<16384, blk, 0, stream>>>(x, xb, TSEQ * HIDDEN);
  tcast_f32_bf16<<<16384, blk, 0, stream>>>(Wq, wqt, HIDDEN, QW);
  tcast_f32_bf16<<<4096,  blk, 0, stream>>>(Wk, wkt, HIDDEN, KVW);
  tcast_f32_bf16<<<4096,  blk, 0, stream>>>(Wv, wvt, HIDDEN, KVW);
  tcast_f32_bf16<<<16384, blk, 0, stream>>>(Wo, wot, QW, HIDDEN);
  // 2) projections (WMMA + async LDS staging)
  gemm_bf16_tn<<<512, blk, 0, stream>>>(xb, wqt, qf, TSEQ, QW,  HIDDEN);
  gemm_bf16_tn<<<128, blk, 0, stream>>>(xb, wkt, kf, TSEQ, KVW, HIDDEN);
  gemm_bf16_tn<<<128, blk, 0, stream>>>(xb, wvt, vf, TSEQ, KVW, HIDDEN);
  // 3) RoPE + V transpose
  rope_cast<<<16384, blk, 0, stream>>>(qf, qr, NH,  rope_scale, pos_offset);
  rope_cast<<<4096,  blk, 0, stream>>>(kf, kr, NKV, rope_scale, pos_offset);
  vtrans<<<4096, blk, 0, stream>>>(vf, vt);
  // 4) causal flash attention (WMMA)
  flash_attn<<<256, blk, 0, stream>>>(qr, kr, vt, attnf);
  // 5) output projection (WMMA)
  cast_f32_bf16<<<16384, blk, 0, stream>>>(attnf, attnb, TSEQ * QW);
  gemm_bf16_tn<<<512, blk, 0, stream>>>(attnb, wot, (float*)d_out, TSEQ, HIDDEN, QW);
}
